// DAbstractor_14405320311561
// MI455X (gfx1250) — matrix operand order repaired and checked by reading, not compiled, not run on hardware
//
#include <hip/hip_runtime.h>
#include <hip/hip_bf16.h>

typedef __attribute__((ext_vector_type(16))) __bf16 v16bf;
typedef __attribute__((ext_vector_type(8)))  __bf16 v8bf;
typedef __attribute__((ext_vector_type(8)))  float  v8f;

#define B_   2
#define C_   1024
#define HW_  4096
#define S_   2048
#define D_   1024
#define NH_  16
#define DH_  64

// LDS tile row stride (bf16 elems): 32 data + 16 pad = 96B rows (16B aligned)
#define TSTRIDE 48

// ---------------------------------------------------------------------------
// WMMA helpers (CDNA5 16x16x32 bf16, f32 accumulate)
// ---------------------------------------------------------------------------
__device__ __forceinline__ v8f wmma_bf16(v16bf a, v16bf b, v8f c) {
  return __builtin_amdgcn_wmma_f32_16x16x32_bf16(false, a, false, b, (short)0, c,
                                                 false, false);
}

// A-matrix 16x32 bf16 fragment (global source), contiguous K order:
//   lane<16 : row=lane,    elems 0..15 = K 0..15
//   lane>=16: row=lane-16, elems 0..15 = K 16..31
__device__ __forceinline__ v16bf load_afrag(const __bf16* __restrict__ A, long lda,
                                            int rowBase, int k0, int lrow, int lhi) {
  const __bf16* p = A + (long)(rowBase + lrow) * lda + (k0 + lhi * 16);
  return *(const v16bf*)(p);
}

// B-matrix 32x16 bf16 fragment, weight-form Bw[N][K] (global source)
__device__ __forceinline__ v16bf load_bfrag(const __bf16* __restrict__ Bw, long ldb,
                                            int colBase, int k0, int lrow, int lhi) {
  const __bf16* p = Bw + (long)(colBase + lrow) * ldb + (k0 + lhi * 16);
  return *(const v16bf*)(p);
}

// Fragment loads from an LDS tile with row stride TSTRIDE
__device__ __forceinline__ v16bf lds_afrag(const __bf16* t, int rowBase,
                                           int lrow, int lhi) {
  return *(const v16bf*)(t + (rowBase + lrow) * TSTRIDE + lhi * 16);
}
__device__ __forceinline__ v16bf lds_bfrag(const __bf16* t, int rowBase,
                                           int lrow, int lhi) {
  return *(const v16bf*)(t + (rowBase + lrow) * TSTRIDE + lhi * 16);
}

// Async global->LDS 16B copy (per lane), tracked by ASYNCcnt.
__device__ __forceinline__ void async_copy16(const void* gaddr, void* lds_ptr) {
  // Low 32 bits of a generic LDS address are the LDS byte offset.
  unsigned lds_off = (unsigned)(unsigned long long)lds_ptr;
  asm volatile("global_load_async_to_lds_b128 %0, %1, off"
               :: "v"(lds_off), "v"((unsigned long long)gaddr)
               : "memory");
}
__device__ __forceinline__ void async_wait_all() {
  asm volatile("s_wait_asynccnt 0x0" ::: "memory");
}

// ---------------------------------------------------------------------------
// Generic GEMM: C[M,N] = A[M,K] @ Bw[N,K]^T + bias
// 256 threads = 8 waves (2x4), block tile 128x128, wave tile 64x32.
// Double-buffered async global->LDS staging of A(128x32) and B(128x32) slabs,
// WMMA consumption from LDS. Output store uses generic strides (supports the
// fused transpose), optional f32 and/or bf16 destinations, batching via z.
// ---------------------------------------------------------------------------
__global__ __launch_bounds__(256) void gemm_bf16_kernel(
    const __bf16* __restrict__ A, long lda, long strideA,
    const __bf16* __restrict__ Bw, long ldb,
    const float* __restrict__ bias,
    float* __restrict__ Cf, __bf16* __restrict__ Cb,
    long c_rs, long c_cs, long strideC,
    int K)
{
  __shared__ __bf16 Abuf[2][128 * TSTRIDE];
  __shared__ __bf16 Bbuf[2][128 * TSTRIDE];

  const int lane = threadIdx.x & 31;
  const int wave = threadIdx.x >> 5;
  const int lrow = lane & 15, lhi = lane >> 4;
  const int wm = wave >> 2;            // 0..1
  const int wn = wave & 3;             // 0..3
  const int blockRow = blockIdx.y * 128;
  const int blockCol = blockIdx.x * 128;
  const __bf16* Ab = A + (long)blockIdx.z * strideA;

  // Cooperative async stage of one 128x32 A slab + 128x32 B slab into buf.
  auto stage = [&](int buf, int kk) {
#pragma unroll
    for (int i = 0; i < 2; ++i) {
      const int c   = threadIdx.x + i * 256;  // 0..511 : 16B chunk id
      const int row = c >> 2;                 // 0..127
      const int cb  = (c & 3) * 16;           // byte offset within 64B row-slab
      const char* ga = (const char*)(Ab + (long)(blockRow + row) * lda + kk) + cb;
      async_copy16(ga, (char*)&Abuf[buf][row * TSTRIDE] + cb);
      const char* gb = (const char*)(Bw + (long)(blockCol + row) * ldb + kk) + cb;
      async_copy16(gb, (char*)&Bbuf[buf][row * TSTRIDE] + cb);
    }
  };

  v8f acc[4][2] = {};

  stage(0, 0);
  async_wait_all();
  __syncthreads();

  int cur = 0;
  for (int k0 = 0; k0 < K; k0 += 32) {
    if (k0 + 32 < K) stage(cur ^ 1, k0 + 32);   // prefetch next slab

    const __bf16* At = Abuf[cur];
    const __bf16* Bt = Bbuf[cur];
    v16bf a[4], b[2];
#pragma unroll
    for (int mi = 0; mi < 4; ++mi)
      a[mi] = lds_afrag(At, wm * 64 + mi * 16, lrow, lhi);
#pragma unroll
    for (int ni = 0; ni < 2; ++ni)
      b[ni] = lds_bfrag(Bt, wn * 32 + ni * 16, lrow, lhi);
#pragma unroll
    for (int mi = 0; mi < 4; ++mi)
#pragma unroll
      for (int ni = 0; ni < 2; ++ni)
        acc[mi][ni] = wmma_bf16(a[mi], b[ni], acc[mi][ni]);

    async_wait_all();
    __syncthreads();
    cur ^= 1;
  }

#pragma unroll
  for (int mi = 0; mi < 4; ++mi)
#pragma unroll
    for (int ni = 0; ni < 2; ++ni) {
      const int col = blockCol + wn * 32 + ni * 16 + lrow;
      const float bv = bias ? bias[col] : 0.0f;
#pragma unroll
      for (int r = 0; r < 8; ++r) {
        const int row = blockRow + wm * 64 + mi * 16 + r + lhi * 8;
        const float v = acc[mi][ni][r] + bv;
        const long off = (long)blockIdx.z * strideC + (long)row * c_rs + (long)col * c_cs;
        if (Cf) Cf[off] = v;
        if (Cb) Cb[off] = (__bf16)v;
      }
    }
}

// ---------------------------------------------------------------------------
// Elementwise f32 -> bf16
// ---------------------------------------------------------------------------
__global__ void f32_to_bf16_kernel(const float* __restrict__ in,
                                   __bf16* __restrict__ out, long n)
{
  long i = ((long)blockIdx.x * blockDim.x + threadIdx.x) * 4;
  if (i + 3 < n) {
    float4 f = *(const float4*)(in + i);
    out[i]     = (__bf16)f.x;
    out[i + 1] = (__bf16)f.y;
    out[i + 2] = (__bf16)f.z;
    out[i + 3] = (__bf16)f.w;
  } else {
    for (long j = i; j < n; ++j) out[j] = (__bf16)in[j];
  }
}

// ---------------------------------------------------------------------------
// LayerNorm over D=1024, one block (256 thr) per row; bf16 out.
// ---------------------------------------------------------------------------
__global__ __launch_bounds__(256) void layernorm_kernel(
    const float* __restrict__ in, const float* __restrict__ g,
    const float* __restrict__ beta, __bf16* __restrict__ out)
{
  const int row = blockIdx.x;
  const int tid = threadIdx.x;
  const int lane = tid & 31, wave = tid >> 5;
  const float* p = in + (long)row * D_;
  float v[4], s = 0.f, ss = 0.f;
#pragma unroll
  for (int i = 0; i < 4; ++i) {
    v[i] = p[tid + i * 256];
    s += v[i]; ss += v[i] * v[i];
  }
#pragma unroll
  for (int off = 16; off >= 1; off >>= 1) {
    s  += __shfl_xor(s, off, 32);
    ss += __shfl_xor(ss, off, 32);
  }
  __shared__ float redS[8], redQ[8];
  if (lane == 0) { redS[wave] = s; redQ[wave] = ss; }
  __syncthreads();
  s = 0.f; ss = 0.f;
#pragma unroll
  for (int wv = 0; wv < 8; ++wv) { s += redS[wv]; ss += redQ[wv]; }
  const float mu  = s * (1.0f / D_);
  const float var = ss * (1.0f / D_) - mu * mu;
  const float inv = rsqrtf(var + 1e-5f);
  __bf16* po = out + (long)row * D_;
#pragma unroll
  for (int i = 0; i < 4; ++i) {
    const int c = tid + i * 256;
    po[c] = (__bf16)((v[i] - mu) * inv * g[c] + beta[c]);
  }
}

// ---------------------------------------------------------------------------
// Q l2-norm: one wave per (b,s,h); in q_f32 (B*S, NH*DH); out (B,NH,S,DH) bf16
// ---------------------------------------------------------------------------
__global__ __launch_bounds__(256) void qnorm_kernel(const float* __restrict__ q,
                                                    __bf16* __restrict__ qo)
{
  const int lane = threadIdx.x & 31, wave = threadIdx.x >> 5;
  const long pair = (long)blockIdx.x * 8 + wave;   // bs*NH + h, < B*S*NH
  const int  h  = (int)(pair & (NH_ - 1));
  const long bs = pair >> 4;
  const float* p = q + bs * (NH_ * DH_) + h * DH_;
  const float e0 = p[lane * 2], e1 = p[lane * 2 + 1];
  float ss = e0 * e0 + e1 * e1;
#pragma unroll
  for (int off = 16; off >= 1; off >>= 1) ss += __shfl_xor(ss, off, 32);
  const float sc = 1.0f / fmaxf(sqrtf(ss), 1e-12f);
  const long b = bs >> 11, s = bs & (S_ - 1);
  __bf16* o = qo + (((b * NH_ + h) * (long)S_) + s) * DH_;
  o[lane * 2]     = (__bf16)(e0 * sc);
  o[lane * 2 + 1] = (__bf16)(e1 * sc);
}

// ---------------------------------------------------------------------------
// KV split: k -> l2norm -> (B,S,DH) bf16 ; v -> transposed (B,DH,S) bf16
// one wave per row (b,s).
// ---------------------------------------------------------------------------
__global__ __launch_bounds__(256) void kvnorm_kernel(const float* __restrict__ kv,
                                                     __bf16* __restrict__ kb,
                                                     __bf16* __restrict__ vt)
{
  const int lane = threadIdx.x & 31, wave = threadIdx.x >> 5;
  const long row = (long)blockIdx.x * 8 + wave;    // < B*S
  const float* p = kv + row * (2 * DH_);
  const float k0 = p[lane * 2], k1 = p[lane * 2 + 1];
  float ss = k0 * k0 + k1 * k1;
#pragma unroll
  for (int off = 16; off >= 1; off >>= 1) ss += __shfl_xor(ss, off, 32);
  const float sc = 1.0f / fmaxf(sqrtf(ss), 1e-12f);
  const long b = row >> 11, s = row & (S_ - 1);
  __bf16* ko = kb + (b * S_ + s) * DH_;
  ko[lane * 2]     = (__bf16)(k0 * sc);
  ko[lane * 2 + 1] = (__bf16)(k1 * sc);
  const float v0 = p[DH_ + lane * 2], v1 = p[DH_ + lane * 2 + 1];
  vt[(b * DH_ + lane * 2) * (long)S_ + s]     = (__bf16)v0;
  vt[(b * DH_ + lane * 2 + 1) * (long)S_ + s] = (__bf16)v1;
}

// ---------------------------------------------------------------------------
// Causal flash attention, one wave per 16 query rows, key blocks of 32.
//   Q  (B,NH,S,DH) bf16 (l2-normed)
//   K  (B,S,DH)    bf16 (l2-normed, shared across heads)
//   Vt (B,DH,S)    bf16 (transposed for contiguous B-fragments)
//   O  (B,S,NH*DH) f32
// ---------------------------------------------------------------------------
__global__ __launch_bounds__(256) void attn_kernel(
    const __bf16* __restrict__ Q,
    const __bf16* __restrict__ Kmat,
    const __bf16* __restrict__ Vt,
    float* __restrict__ O)
{
  __shared__ __bf16 pbuf[8][16 * TSTRIDE];   // per-wave 16x32 P tile
  const int lane = threadIdx.x & 31;
  const int wave = threadIdx.x >> 5;
  const int lrow = lane & 15, lhi = lane >> 4;
  const int bh = blockIdx.y, b = bh >> 4, h = bh & 15;
  const int q0 = blockIdx.x * 128 + wave * 16;

  const __bf16* qp = Q + ((long)bh * S_ + q0) * DH_;
  const __bf16* kp = Kmat + (long)b * S_ * DH_;
  const __bf16* vp = Vt + (long)b * DH_ * S_;
  __bf16* pl = pbuf[wave];

  v16bf aq[2];
#pragma unroll
  for (int ks = 0; ks < 2; ++ks)
    aq[ks] = load_afrag(qp, DH_, 0, ks * 32, lrow, lhi);

  v8f o[4] = {};
  float m[8], l[8];
#pragma unroll
  for (int r = 0; r < 8; ++r) { m[r] = -3.0e38f; l[r] = 0.0f; }

  for (int kb = 0; kb < q0 + 16; kb += 32) {
    // sim = (q . k) over DH=64, two 16-key N-tiles
    v8f st[2] = {};
#pragma unroll
    for (int t = 0; t < 2; ++t)
#pragma unroll
      for (int ks = 0; ks < 2; ++ks) {
        v16bf bk = load_bfrag(kp, DH_, kb + t * 16, ks * 32, lrow, lhi);
        st[t] = wmma_bf16(aq[ks], bk, st[t]);
      }

    float alpha[8];
#pragma unroll
    for (int r = 0; r < 8; ++r) {
      const int qi = q0 + r + lhi * 8;
      float s0 = st[0][r] * 8.0f;                 // QK_SCALE
      float s1 = st[1][r] * 8.0f;
      if (kb + lrow > qi)      s0 = -3.0e38f;     // causal mask
      if (kb + 16 + lrow > qi) s1 = -3.0e38f;
      float mx = fmaxf(s0, s1);
#pragma unroll
      for (int off = 8; off >= 1; off >>= 1)
        mx = fmaxf(mx, __shfl_xor(mx, off, 32));  // row reduce (16-lane groups)
      const float mn = fmaxf(m[r], mx);
      alpha[r] = __expf(m[r] - mn);
      const float p0 = __expf(s0 - mn);
      const float p1 = __expf(s1 - mn);
      float ps = p0 + p1;
#pragma unroll
      for (int off = 8; off >= 1; off >>= 1) ps += __shfl_xor(ps, off, 32);
      l[r] = l[r] * alpha[r] + ps;
      m[r] = mn;
      // C-layout -> LDS (wave-private; DS ops are in-order within a wave)
      pl[(r + lhi * 8) * TSTRIDE + lrow]      = (__bf16)p0;
      pl[(r + lhi * 8) * TSTRIDE + 16 + lrow] = (__bf16)p1;
    }

#pragma unroll
    for (int nt = 0; nt < 4; ++nt)
#pragma unroll
      for (int r = 0; r < 8; ++r) o[nt][r] *= alpha[r];

    // reload P as an A-fragment (16x32, K = 32 keys)
    v16bf ap = lds_afrag(pl, 0, lrow, lhi);

    // O += P @ V  (4 N-tiles over DH=64)
#pragma unroll
    for (int nt = 0; nt < 4; ++nt) {
      const __bf16* vb = vp + (long)(nt * 16 + lrow) * S_ + kb + lhi * 16;
      v16bf bv = *(const v16bf*)vb;
      o[nt] = wmma_bf16(ap, bv, o[nt]);
    }
  }

#pragma unroll
  for (int nt = 0; nt < 4; ++nt)
#pragma unroll
    for (int r = 0; r < 8; ++r) {
      const int row = q0 + r + lhi * 8;
      O[((long)b * S_ + row) * (NH_ * DH_) + h * DH_ + nt * 16 + lrow] =
          o[nt][r] / l[r];
    }
}

// ---------------------------------------------------------------------------
// Host launcher
// ---------------------------------------------------------------------------
extern "C" void kernel_launch(void* const* d_in, const int* in_sizes, int n_in,
                              void* d_out, int out_size, void* d_ws, size_t ws_size,
                              hipStream_t stream) {
  (void)in_sizes; (void)n_in; (void)out_size; (void)ws_size;
  const float* x    = (const float*)d_in[0];
  const float* w_hw = (const float*)d_in[1];
  const float* b_hw = (const float*)d_in[2];
  const float* w_c  = (const float*)d_in[3];
  const float* b_c  = (const float*)d_in[4];
  const float* ln_g = (const float*)d_in[5];
  const float* ln_b = (const float*)d_in[6];
  const float* wq   = (const float*)d_in[7];
  const float* wkv  = (const float*)d_in[8];
  const float* wo   = (const float*)d_in[9];
  float* out = (float*)d_out;

  char* wsp = (char*)d_ws;
  auto alloc = [&](size_t bytes) -> void* {
    void* p = (void*)wsp;
    wsp += (bytes + 255) & ~(size_t)255;
    return p;
  };

  const long n_x    = (long)B_ * C_ * HW_;      // 8388608
  const long n_whw  = (long)S_ * HW_;           // 8388608
  const long n_wc   = (long)D_ * C_;            // 1048576
  const long n_wq   = (long)NH_ * DH_ * D_;     // 1048576
  const long n_wkv  = (long)2 * DH_ * D_;       // 131072
  const long n_wo   = (long)D_ * NH_ * DH_;     // 1048576
  const long n_img  = (long)B_ * S_ * D_;       // 4194304
  const long n_kv   = (long)B_ * S_ * 2 * DH_;  // 524288
  const long n_kd   = (long)B_ * S_ * DH_;      // 262144

  __bf16* x_b    = (__bf16*)alloc(n_x   * 2);
  __bf16* whw_b  = (__bf16*)alloc(n_whw * 2);
  __bf16* wc_b   = (__bf16*)alloc(n_wc  * 2);
  __bf16* wq_b   = (__bf16*)alloc(n_wq  * 2);
  __bf16* wkv_b  = (__bf16*)alloc(n_wkv * 2);
  __bf16* wo_b   = (__bf16*)alloc(n_wo  * 2);
  __bf16* img1_b = (__bf16*)alloc(n_img * 2);   // stage1 out, (B,S,C) bf16
  float*  img2_f = (float*) alloc(n_img * 4);   // stage2 out pre-LN (B,S,D)
  __bf16* img_b  = (__bf16*)alloc(n_img * 2);   // post-LN (B,S,D) bf16
  float*  q_f    = (float*) alloc(n_img * 4);   // q pre-norm (B*S, NH*DH)
  float*  kv_f   = (float*) alloc(n_kv  * 4);   // kv (B*S, 2*DH)
  __bf16* q_b    = (__bf16*)alloc(n_img * 2);   // (B,NH,S,DH)
  __bf16* k_b    = (__bf16*)alloc(n_kd  * 2);   // (B,S,DH)
  __bf16* vt_b   = (__bf16*)alloc(n_kd  * 2);   // (B,DH,S)
  float*  ao_f   = (float*) alloc(n_img * 4);   // attention out (B,S,NH*DH)
  __bf16* ao_b   = (__bf16*)alloc(n_img * 2);

  const dim3 thr(256);
  auto cvt = [&](const float* src, __bf16* dst, long n) {
    f32_to_bf16_kernel<<<dim3((unsigned)(n / 1024)), thr, 0, stream>>>(src, dst, n);
  };

  // 0. f32 -> bf16 for all matrix operands
  cvt(x, x_b, n_x);
  cvt(w_hw, whw_b, n_whw);
  cvt(w_c, wc_b, n_wc);
  cvt(wq, wq_b, n_wq);
  cvt(wkv, wkv_b, n_wkv);
  cvt(wo, wo_b, n_wo);

  // 1. img1[b,s,c] = x[b,c,:] . w_hw[s,:] + b_hw[s]   (transposed store)
  gemm_bf16_kernel<<<dim3(S_ / 128, C_ / 128, B_), thr, 0, stream>>>(
      x_b, HW_, (long)C_ * HW_, whw_b, HW_, b_hw,
      nullptr, img1_b, /*c_rs=*/1, /*c_cs=*/C_, /*strideC=*/(long)S_ * C_, HW_);

  // 2. img2[b,s,d] = img1[b,s,:] . w_c[d,:] + b_c[d]
  gemm_bf16_kernel<<<dim3(D_ / 128, (B_ * S_) / 128, 1), thr, 0, stream>>>(
      img1_b, C_, 0, wc_b, C_, b_c,
      img2_f, nullptr, D_, 1, 0, C_);

  // 3. LayerNorm -> bf16
  layernorm_kernel<<<dim3(B_ * S_), thr, 0, stream>>>(img2_f, ln_g, ln_b, img_b);

  // 4. q = img @ wq.T ; kv = img @ wkv.T
  gemm_bf16_kernel<<<dim3((NH_ * DH_) / 128, (B_ * S_) / 128, 1), thr, 0, stream>>>(
      img_b, D_, 0, wq_b, D_, nullptr, q_f, nullptr, NH_ * DH_, 1, 0, D_);
  gemm_bf16_kernel<<<dim3((2 * DH_) / 128, (B_ * S_) / 128, 1), thr, 0, stream>>>(
      img_b, D_, 0, wkv_b, D_, nullptr, kv_f, nullptr, 2 * DH_, 1, 0, D_);

  // 5. l2-norm q/k, layout for attention (v transposed)
  qnorm_kernel<<<dim3((B_ * S_ * NH_) / 8), thr, 0, stream>>>(q_f, q_b);
  kvnorm_kernel<<<dim3((B_ * S_) / 8), thr, 0, stream>>>(kv_f, k_b, vt_b);

  // 6. causal flash attention
  attn_kernel<<<dim3(S_ / 128, B_ * NH_), thr, 0, stream>>>(q_b, k_b, vt_b, ao_f);

  // 7. final projection: out = attn_out @ wo.T
  cvt(ao_f, ao_b, n_img);
  gemm_bf16_kernel<<<dim3(D_ / 128, (B_ * S_) / 128, 1), thr, 0, stream>>>(
      ao_b, NH_ * DH_, 0, wo_b, NH_ * DH_, nullptr, out, nullptr, D_, 1, 0, NH_ * DH_);
}
